// Entropy_55525337203040
// MI455X (gfx1250) — matrix operand order repaired and checked by reading, not compiled, not run on hardware
//
#include <hip/hip_runtime.h>

// Problem constants: NQ=256 queries, NG=20000 gallery, D=2048, K=32
#define NQ   256
#define NG   20000
#define DIM  2048
#define KSEL 32

// GEMM tiling
#define BN    80          // gallery rows per block (NG/BN = 250 exactly)
#define NSUB  5           // 16-wide N subtiles per wave (BN/16)
#define KC    64          // K chunk staged in LDS (2 WMMA k-steps)
#define BSTR  72          // LDS row stride in ushorts (64 + 8 pad -> bank spread)

typedef __attribute__((ext_vector_type(16))) __bf16 v16bf;
typedef __attribute__((ext_vector_type(8)))  float  v8f;

union BFrag {
    v16bf bf;
    unsigned short us[16];
};

// Split fp32 into bf16 hi + bf16 lo (truncation); x ~= hi + lo to ~2^-16 rel.
__device__ __forceinline__ void split_bf16(float x, unsigned short& hi, unsigned short& lo) {
    unsigned int b = __float_as_uint(x);
    hi = (unsigned short)(b >> 16);
    float hf = __uint_as_float(b & 0xFFFF0000u);
    float l  = x - hf;
    lo = (unsigned short)(__float_as_uint(l) >> 16);
}

// A fragment (16-bit A layout): halves 0..7 <- p[0..7], halves 8..15 <- p[16..23]
__device__ __forceinline__ void ldfragA(const unsigned short* p, BFrag& f) {
    *(uint4*)&f.us[0] = *(const uint4*)(p);
    *(uint4*)&f.us[8] = *(const uint4*)(p + 16);
}

// B fragment from LDS: halves 0..7 <- p[0..7], halves 8..15 <- p[16..23]
__device__ __forceinline__ void ldfragB(const unsigned short* p, BFrag& f) {
    *(uint4*)&f.us[0] = *(const uint4*)(p);
    *(uint4*)&f.us[8] = *(const uint4*)(p + 16);
}

#define WMMA_BF16(A, B, C) \
    __builtin_amdgcn_wmma_f32_16x16x32_bf16(false, (A).bf, false, (B).bf, (short)0, (C), false, false)

// ---------------------------------------------------------------------------
// Kernel 0: feat fp32 -> packed bf16 hi/lo arrays (runs once, 0.5M elements)
// ---------------------------------------------------------------------------
__global__ __launch_bounds__(256) void feat_cvt_kernel(const float* __restrict__ feat,
                                                       unsigned short* __restrict__ Ahi,
                                                       unsigned short* __restrict__ Alo) {
    const int i = blockIdx.x * 256 + threadIdx.x;   // grid covers NQ*DIM
    split_bf16(feat[i], Ahi[i], Alo[i]);
}

// ---------------------------------------------------------------------------
// Kernel 1: g2[n] = sum_k gallery[n,k]^2  (also warms gallery into 192MB L2)
// ---------------------------------------------------------------------------
__global__ __launch_bounds__(256) void g2_kernel(const float* __restrict__ g,
                                                 float* __restrict__ g2) {
    __shared__ float red[256];
    const int n = blockIdx.x;
    const float* row = g + (size_t)n * DIM;
    float s = 0.f;
    for (int i = threadIdx.x; i < DIM; i += 256) {
        float v = row[i];
        s += v * v;
    }
    red[threadIdx.x] = s;
    __syncthreads();
    for (int off = 128; off > 0; off >>= 1) {
        if (threadIdx.x < off) red[threadIdx.x] += red[threadIdx.x + off];
        __syncthreads();
    }
    if (threadIdx.x == 0) g2[n] = red[0];
}

// ---------------------------------------------------------------------------
// Kernel 2: dist[m,n] = g2[n] - 2 * dot(feat[m,:], gallery[n,:])
// (||q||^2 dropped: per-row constant -> no effect on top-k or softmax entropy)
// Split-bf16 WMMA: acc += Ahi*Bhi + Ahi*Blo + Alo*Bhi (f32 accumulate).
// Block = 256 thr = 8 waves. Block tile 256(M) x 80(N); wave w: rows [32w,32w+32)
// x all 80 cols (2 M-subtiles x 5 N-subtiles = 10 accumulators).
// B tile converted once per block into LDS; A read as pre-packed bf16 from L2.
// ---------------------------------------------------------------------------
__global__ __launch_bounds__(256) void gemm_dist_kernel(const unsigned short* __restrict__ Ahi,
                                                        const unsigned short* __restrict__ Alo,
                                                        const float* __restrict__ gallery,
                                                        const float* __restrict__ g2,
                                                        float* __restrict__ dist) {
    __shared__ unsigned short Bhi[BN * BSTR];   // 11520 B
    __shared__ unsigned short Blo[BN * BSTR];   // 11520 B

    const int tid   = threadIdx.x;
    const int lane  = tid & 31;
    const int wave  = tid >> 5;
    const int lhalf = lane >> 4;    // 0 or 1
    const int lmod  = lane & 15;
    const int nbase = blockIdx.x * BN;

    // Wave's A rows: subtile0 row = 32w + lane%16, subtile1 = +16
    const unsigned short* A0h = Ahi + (size_t)(wave * 32 + lmod) * DIM;
    const unsigned short* A0l = Alo + (size_t)(wave * 32 + lmod) * DIM;
    const int akoff = lhalf * 8;

    v8f acc[2][NSUB] = {};

    for (int k0 = 0; k0 < DIM; k0 += KC) {
        __syncthreads();   // previous chunk's compute done before LDS overwrite

        // Cooperative stage: 80 rows x 64 K fp32 = 1280 float4, 5 per thread.
        // Threads t..t+15 read 256B contiguous of one gallery row (coalesced).
#pragma unroll
        for (int j = 0; j < 5; ++j) {
            const int idx = tid + 256 * j;        // 0..1279
            const int row = idx >> 4;             // 0..79
            const int kc4 = (idx & 15) << 2;      // 0,4,...,60
            const float* gp = gallery + (size_t)(nbase + row) * DIM + k0 + kc4;
            __builtin_prefetch(gp + KC, 0, 3);    // next chunk (L2-resident stream)
            const float4 c = *(const float4*)gp;
            unsigned short* ph = &Bhi[row * BSTR + kc4];
            unsigned short* pl = &Blo[row * BSTR + kc4];
            split_bf16(c.x, ph[0], pl[0]);
            split_bf16(c.y, ph[1], pl[1]);
            split_bf16(c.z, ph[2], pl[2]);
            split_bf16(c.w, ph[3], pl[3]);
        }
        __syncthreads();

#pragma unroll
        for (int ks = 0; ks < 2; ++ks) {
            const int kk = k0 + ks * 32;
            BFrag a0h, a0l, a1h, a1l;
            ldfragA(A0h + kk + akoff, a0h);
            ldfragA(A0l + kk + akoff, a0l);
            ldfragA(A0h + (size_t)16 * DIM + kk + akoff, a1h);
            ldfragA(A0l + (size_t)16 * DIM + kk + akoff, a1l);

#pragma unroll
            for (int ns = 0; ns < NSUB; ++ns) {
                BFrag bh, bl;
                const int roff = (ns * 16 + lmod) * BSTR + ks * 32 + lhalf * 8;
                ldfragB(&Bhi[roff], bh);
                ldfragB(&Blo[roff], bl);
                acc[0][ns] = WMMA_BF16(a0h, bh, acc[0][ns]);
                acc[0][ns] = WMMA_BF16(a0h, bl, acc[0][ns]);
                acc[0][ns] = WMMA_BF16(a0l, bh, acc[0][ns]);
                acc[1][ns] = WMMA_BF16(a1h, bh, acc[1][ns]);
                acc[1][ns] = WMMA_BF16(a1h, bl, acc[1][ns]);
                acc[1][ns] = WMMA_BF16(a1l, bh, acc[1][ns]);
            }
        }
    }

    // C/D layout: VGPR r -> M = r + 8*(lane>=16); N = lane%16
#pragma unroll
    for (int ns = 0; ns < NSUB; ++ns) {
        const int n  = nbase + ns * 16 + lmod;
        const float gv = g2[n];
#pragma unroll
        for (int r = 0; r < 8; ++r) {
            const int m0 = wave * 32 + r + lhalf * 8;
            dist[(size_t)m0 * NG + n]        = gv - 2.0f * acc[0][ns][r];
            dist[(size_t)(m0 + 16) * NG + n] = gv - 2.0f * acc[1][ns][r];
        }
    }
}

// ---------------------------------------------------------------------------
// Kernel 3: per query row, 32 smallest distances -> softmax entropy.
// Row cached in 80KB LDS (WGP has 320KB). Selection = 32 deterministic
// lexicographic (value,idx) argmin passes via orderable-u64 keys.
// ---------------------------------------------------------------------------
__global__ __launch_bounds__(256) void topk_entropy_kernel(const float* __restrict__ dist,
                                                           float* __restrict__ ent) {
    __shared__ float rowbuf[NG];                 // 80,000 B
    __shared__ unsigned long long red[256];
    __shared__ float selv[KSEL];

    const int m = blockIdx.x;
    const float* row = dist + (size_t)m * NG;
    for (int i = threadIdx.x; i < NG; i += 256) rowbuf[i] = row[i];
    __syncthreads();

    unsigned long long last = 0ull;
    for (int it = 0; it < KSEL; ++it) {
        unsigned long long best = ~0ull;
        for (int i = threadIdx.x; i < NG; i += 256) {
            float v = rowbuf[i];
            unsigned int b = __float_as_uint(v);
            unsigned int o = (b & 0x80000000u) ? ~b : (b | 0x80000000u);  // order-preserving
            unsigned long long key = ((unsigned long long)o << 32) | (unsigned int)i;
            if (key > last && key < best) best = key;
        }
        red[threadIdx.x] = best;
        __syncthreads();
        for (int off = 128; off > 0; off >>= 1) {
            if (threadIdx.x < off) {
                unsigned long long o2 = red[threadIdx.x + off];
                if (o2 < red[threadIdx.x]) red[threadIdx.x] = o2;
            }
            __syncthreads();
        }
        last = red[0];
        if (threadIdx.x == 0) {
            unsigned int o  = (unsigned int)(last >> 32);
            unsigned int bb = (o & 0x80000000u) ? (o & 0x7FFFFFFFu) : ~o;
            selv[it] = __uint_as_float(bb);
        }
        __syncthreads();
    }

    if (threadIdx.x == 0) {
        // logits l_i = -d_i; H = log(sum e^x) - sum(x e^x)/sum(e^x), x = l - lmax
        float lmax = -selv[0];   // selv ascending -> selv[0] = min distance
        float s = 0.f, t = 0.f;
        for (int i = 0; i < KSEL; ++i) {
            float x = -selv[i] - lmax;
            float e = expf(x);
            s += e;
            t += x * e;
        }
        ent[m] = logf(s) - t / s;
    }
}

// ---------------------------------------------------------------------------
// Kernel 4: mean over 256 entropies
// ---------------------------------------------------------------------------
__global__ __launch_bounds__(256) void mean_kernel(const float* __restrict__ ent,
                                                   float* __restrict__ out) {
    __shared__ float red[256];
    red[threadIdx.x] = ent[threadIdx.x];
    __syncthreads();
    for (int off = 128; off > 0; off >>= 1) {
        if (threadIdx.x < off) red[threadIdx.x] += red[threadIdx.x + off];
        __syncthreads();
    }
    if (threadIdx.x == 0) out[0] = red[0] * (1.0f / (float)NQ);
}

// ---------------------------------------------------------------------------
extern "C" void kernel_launch(void* const* d_in, const int* in_sizes, int n_in,
                              void* d_out, int out_size, void* d_ws, size_t ws_size,
                              hipStream_t stream) {
    const float* feat    = (const float*)d_in[0];   // [256, 2048] fp32
    const float* gallery = (const float*)d_in[1];   // [20000, 2048] fp32
    // d_in[2] is k (==32), fixed at compile time.

    char* base = (char*)d_ws;
    float*          g2   = (float*)base;                           // 20000 f (80,000 B)
    unsigned short* Ahi  = (unsigned short*)(base + 81920);        // 524288 us (1 MB)
    unsigned short* Alo  = (unsigned short*)(base + 81920 + 1048576);
    float*          dist = (float*)(base + 81920 + 2097152);       // 256*20000 f
    float*          ent  = dist + (size_t)NQ * NG;                 // 256 f
    float*          out  = (float*)d_out;

    feat_cvt_kernel<<<(NQ * DIM) / 256, 256, 0, stream>>>(feat, Ahi, Alo);
    g2_kernel<<<NG, 256, 0, stream>>>(gallery, g2);
    gemm_dist_kernel<<<NG / BN, 256, 0, stream>>>(Ahi, Alo, gallery, g2, dist);
    topk_entropy_kernel<<<NQ, 256, 0, stream>>>(dist, ent);
    mean_kernel<<<1, 256, 0, stream>>>(ent, out);
}
